// Attention_20375324852589
// MI455X (gfx1250) — compile-verified, matrix operand order checked by reading, not run on hardware
//
#include <hip/hip_runtime.h>
#include <hip/hip_bf16.h>
#include <cstdint>

#define B_    64
#define T_    2048
#define RNN_  1024
#define EMB_  512
#define ATTN_ 128
#define NF_   32
#define KS_   31
#define PAD_  15

typedef __attribute__((ext_vector_type(16))) _Float16 v16h;
typedef __attribute__((ext_vector_type(8)))  float    v8f;

// Branch-free tanh: copysign(1 - 2/(exp2(2*log2e*|x|)+1), x).
// v_exp_f32 + v_rcp_f32, no EXEC manipulation, no spills; saturates to +-1.
__device__ __forceinline__ float fast_tanh(float x) {
    float ax = __builtin_fabsf(x);
    float e2 = __builtin_amdgcn_exp2f(ax * 2.885390081777926f); // 2*log2(e)
    float r  = 1.0f - 2.0f * __builtin_amdgcn_rcpf(e2 + 1.0f);
    return __builtin_copysignf(r, x);
}

// exp(x) for x <= 0 (softmax after max-subtraction): single v_exp_f32.
__device__ __forceinline__ float fast_exp_nonpos(float x) {
    return __builtin_amdgcn_exp2f(x * 1.4426950408889634f);
}

// ---------------------------------------------------------------------------
// Kernel 1: pq[b,a] = sum_r hidden[b,r] * Wq[a,r]
// 64 blocks x 128 threads; hidden row staged in LDS, Wq stays in L2 (512 KB).
// ---------------------------------------------------------------------------
__global__ void pq_kernel(const float* __restrict__ hidden,
                          const float* __restrict__ Wq,
                          float* __restrict__ pq) {
    __shared__ float h[RNN_];
    const int b   = blockIdx.x;
    const int tid = threadIdx.x;                 // 0..127 == attn dim
    for (int i = tid; i < RNN_; i += 128) h[i] = hidden[b * RNN_ + i];
    __syncthreads();
    const float* wrow = Wq + (size_t)tid * RNN_;
    float s = 0.0f;
    for (int r = 0; r < RNN_; ++r) s += h[r] * wrow[r];
    pq[b * ATTN_ + tid] = s;
}

// ---------------------------------------------------------------------------
// Kernel 2: fused conv + ploc (WMMA) + tanh + Wv dot -> energies[b,t]
// One block per (batch, 128-wide T tile): grid = 64*16, 256 threads (8 waves).
// Each wave owns 16 t-rows; ploc via v_wmma_f32_16x16x32_f16, C carries
// pq + processed_memory in exact f32.
// ---------------------------------------------------------------------------
__global__ void energies_kernel(const float* __restrict__ aw,      // [B,2,T]
                                const float* __restrict__ Wconv,   // [32,2,31]
                                const float* __restrict__ Wloc,    // [128,32]
                                const float* __restrict__ Wv,      // [1,128]
                                const float* __restrict__ pq,      // [B,128]
                                const float* __restrict__ pm,      // [B,T,128]
                                float* __restrict__ energies) {    // [B,T]
    __shared__ float    awLds[2][128 + KS_ - 1];   // 2 x 158
    __shared__ float    WconvLds[NF_ * 2 * KS_];   // 1984 f32
    __shared__ float    convLds[NF_ * 128];        // [f][t_local], 16 KB
    __shared__ _Float16 WlocH[ATTN_ * NF_];        // f16 copy of Wloc
    __shared__ float    pqLds[ATTN_];
    __shared__ float    WvLds[ATTN_];

    const int blk  = blockIdx.x;
    const int b    = blk >> 4;       // 16 tiles per batch
    const int tile = blk & 15;
    const int t0   = tile * 128;
    const int tid  = threadIdx.x;

    // ---- phase 1: stage inputs ----
    for (int i = tid; i < 2 * (128 + KS_ - 1); i += 256) {
        int c = i / (128 + KS_ - 1);
        int j = i - c * (128 + KS_ - 1);
        int t = t0 + j - PAD_;
        awLds[c][j] = (t >= 0 && t < T_) ? aw[((size_t)b * 2 + c) * T_ + t] : 0.0f;
    }
    for (int i = tid; i < NF_ * 2 * KS_; i += 256) WconvLds[i] = Wconv[i];
    for (int i = tid; i < ATTN_ * NF_;   i += 256) WlocH[i] = (_Float16)Wloc[i];
    if (tid < ATTN_) { pqLds[tid] = pq[b * ATTN_ + tid]; WvLds[tid] = Wv[tid]; }
    __syncthreads();

    // ---- phase 2: 1-D conv (cross-correlation, matching lax.conv) ----
    for (int idx = tid; idx < NF_ * 128; idx += 256) {
        int f  = idx >> 7;
        int tl = idx & 127;
        float s = 0.0f;
        #pragma unroll
        for (int c = 0; c < 2; ++c) {
            const float* wrow = &WconvLds[(f * 2 + c) * KS_];
            const float* arow = &awLds[c][tl];
            #pragma unroll
            for (int k = 0; k < KS_; ++k) s += arow[k] * wrow[k];
        }
        convLds[f * 128 + tl] = s;
    }
    __syncthreads();

    // ---- phase 3: per-wave WMMA over attn dim ----
    const int wave = tid >> 5;
    const int lane = tid & 31;
    const int tl0  = wave * 16;       // this wave's local t base
    const int Mi   = lane & 15;       // row (A) / col (B,C) index
    const int half = lane >> 4;

    // A fragment: 16x32 f16, ISA 7.12.2 layout
    v16h afrag;
    #pragma unroll
    for (int v = 0; v < 8; ++v) {
        int kb = (v < 4) ? (half * 8 + 2 * v) : (16 + half * 8 + 2 * (v - 4));
        afrag[2 * v]     = (_Float16)convLds[kb * 128 + tl0 + Mi];
        afrag[2 * v + 1] = (_Float16)convLds[(kb + 1) * 128 + tl0 + Mi];
    }

    float eacc[8];
    #pragma unroll
    for (int v = 0; v < 8; ++v) eacc[v] = 0.0f;

    // unroll 4 (not 8): halves live B-fragments/C-tiles -> no scratch spills,
    // still 4 independent WMMAs in flight for the scheduler.
    #pragma unroll 4
    for (int n = 0; n < 8; ++n) {            // 8 N-chunks of 16 attn dims
        const int a = n * 16 + Mi;
        // B fragment: B[k, a] = Wloc[a, k], mirrored 32x16 f16 layout
        v16h bfrag;
        #pragma unroll
        for (int v = 0; v < 8; ++v) {
            int kb = (v < 4) ? (half * 8 + 2 * v) : (16 + half * 8 + 2 * (v - 4));
            bfrag[2 * v]     = WlocH[a * NF_ + kb];
            bfrag[2 * v + 1] = WlocH[a * NF_ + kb + 1];
        }
        const float pq_a = pqLds[a];
        const float wv_a = WvLds[a];
        // C = pq + processed_memory (exact f32 accumulate)
        v8f cacc;
        #pragma unroll
        for (int v = 0; v < 8; ++v) {
            int t = t0 + tl0 + v + half * 8;
            cacc[v] = pq_a + pm[((size_t)b * T_ + t) * ATTN_ + a];
        }
        v8f dres = __builtin_amdgcn_wmma_f32_16x16x32_f16(
            /*neg_a=*/false, afrag, /*neg_b=*/false, bfrag,
            /*c_mod=*/(short)0, cacc, /*reuse_a=*/false, /*reuse_b=*/false);
        #pragma unroll
        for (int v = 0; v < 8; ++v) eacc[v] += fast_tanh(dres[v]) * wv_a;
    }

    // cross-lane reduce over the 16-lane halves (matches C/D layout)
    #pragma unroll
    for (int v = 0; v < 8; ++v) {
        float s = eacc[v];
        s += __shfl_xor(s, 1);
        s += __shfl_xor(s, 2);
        s += __shfl_xor(s, 4);
        s += __shfl_xor(s, 8);
        if (Mi == 0) {
            int t = t0 + tl0 + v + half * 8;
            energies[(size_t)b * T_ + t] = s;
        }
    }
}

// ---------------------------------------------------------------------------
// Kernel 3: masked softmax over T, one block per batch.
// ---------------------------------------------------------------------------
__global__ void softmax_kernel(const float* __restrict__ energies,
                               const uint8_t* __restrict__ mask,
                               float* __restrict__ weights) {
    __shared__ float red[256];
    const int b   = blockIdx.x;
    const int tid = threadIdx.x;
    const float NEG = -__builtin_inff();

    float m = NEG;
    for (int t = tid; t < T_; t += 256) {
        float e = mask[b * T_ + t] ? NEG : energies[b * T_ + t];
        m = fmaxf(m, e);
    }
    red[tid] = m; __syncthreads();
    for (int s = 128; s > 0; s >>= 1) {
        if (tid < s) red[tid] = fmaxf(red[tid], red[tid + s]);
        __syncthreads();
    }
    const float mx = red[0];
    __syncthreads();

    float sum = 0.0f;
    for (int t = tid; t < T_; t += 256) {
        float e = mask[b * T_ + t] ? NEG : energies[b * T_ + t];
        sum += fast_exp_nonpos(e - mx);     // e <= mx, never overflows
    }
    red[tid] = sum; __syncthreads();
    for (int s = 128; s > 0; s >>= 1) {
        if (tid < s) red[tid] += red[tid + s];
        __syncthreads();
    }
    const float inv = __builtin_amdgcn_rcpf(red[0]);

    for (int t = tid; t < T_; t += 256) {
        float e = mask[b * T_ + t] ? NEG : energies[b * T_ + t];
        weights[b * T_ + t] = fast_exp_nonpos(e - mx) * inv;
    }
}

// ---------------------------------------------------------------------------
// Kernel 4: context[b,e] = sum_t w[b,t] * memory[b,t,e]
// grid = 64 batches x 8 T-chunks; coalesced streaming of memory (268 MB),
// f32 atomic partial sums (context region pre-zeroed via memset).
// ---------------------------------------------------------------------------
__global__ void context_kernel(const float* __restrict__ memory,
                               const float* __restrict__ weights,
                               float* __restrict__ ctx) {
    const int bc    = blockIdx.x;
    const int b     = bc >> 3;
    const int chunk = bc & 7;
    const int t0    = chunk * (T_ / 8);
    const int e     = threadIdx.x;               // 0..255
    float a0 = 0.0f, a1 = 0.0f;
    for (int t = t0; t < t0 + (T_ / 8); ++t) {
        const float  w    = weights[b * T_ + t];
        const float* mrow = memory + ((size_t)b * T_ + t) * EMB_;
        a0 += w * mrow[e];
        a1 += w * mrow[e + 256];
    }
    atomicAdd(&ctx[b * EMB_ + e],       a0);
    atomicAdd(&ctx[b * EMB_ + e + 256], a1);
}

// ---------------------------------------------------------------------------
extern "C" void kernel_launch(void* const* d_in, const int* in_sizes, int n_in,
                              void* d_out, int out_size, void* d_ws, size_t ws_size,
                              hipStream_t stream) {
    (void)in_sizes; (void)n_in; (void)out_size; (void)ws_size;

    const float*   hidden = (const float*)d_in[0];   // [64,1024]
    const float*   memory = (const float*)d_in[1];   // [64,2048,512]
    const float*   pm     = (const float*)d_in[2];   // [64,2048,128]
    const float*   awcat  = (const float*)d_in[3];   // [64,2,2048]
    const uint8_t* mask   = (const uint8_t*)d_in[4]; // [64,2048] bool
    const float*   Wq     = (const float*)d_in[5];   // [128,1024]
    const float*   Wconv  = (const float*)d_in[6];   // [32,2,31]
    const float*   Wloc   = (const float*)d_in[7];   // [128,32]
    const float*   Wv     = (const float*)d_in[8];   // [1,128]

    float* out     = (float*)d_out;
    float* ctx     = out;               // [64,512]  (output 0)
    float* weights = out + B_ * EMB_;   // [64,2048] (output 1)

    float* pq_ws    = (float*)d_ws;             // 64*128 f32
    float* energies = pq_ws + B_ * ATTN_;       // 64*2048 f32

    pq_kernel<<<B_, 128, 0, stream>>>(hidden, Wq, pq_ws);
    energies_kernel<<<B_ * (T_ / 128), 256, 0, stream>>>(
        awcat, Wconv, Wloc, Wv, pq_ws, pm, energies);
    softmax_kernel<<<B_, 256, 0, stream>>>(energies, mask, weights);
    hipMemsetAsync(ctx, 0, (size_t)B_ * EMB_ * sizeof(float), stream);
    context_kernel<<<B_ * 8, 256, 0, stream>>>(memory, weights, ctx);
}
